// COBRABlock_89850715832422
// MI455X (gfx1250) — compile-verified, matrix operand order checked by reading, not run on hardware
//
#include <hip/hip_runtime.h>
#include <math.h>

// ---------------------------------------------------------------------------
// COBRA block (dual Mamba + FFN) for gfx1250.
// GEMMs: bf16 WMMA (v_wmma_f32_16x16x32_bf16), fp32 accumulate, double-
// buffered LDS 64x64 block tiles with register-staged software pipeline;
// B staged transposed so fragments load as ds_load_b128.
// Scan: wave-parallel over channels, 16 lanes = 16 SSM states.
// ---------------------------------------------------------------------------

typedef unsigned short u16;
typedef __attribute__((ext_vector_type(16))) __bf16 v16bf;
typedef __attribute__((ext_vector_type(16))) u16   v16u;
typedef __attribute__((ext_vector_type(8)))  float v8f;

__device__ __forceinline__ u16 f2bf(float f) {
  unsigned u = __float_as_uint(f);
  unsigned r = u + 0x7FFFu + ((u >> 16) & 1u);   // round-to-nearest-even
  return (u16)(r >> 16);
}

__device__ __forceinline__ float silu_f(float x) {
  return x / (1.f + __expf(-x));
}

enum { ACT_NONE = 0, ACT_SOFTPLUS = 1, ACT_GELU = 2 };

template <int ACT>
__device__ __forceinline__ float apply_act(float x) {
  if (ACT == ACT_SOFTPLUS) return (x > 20.f) ? x : log1pf(__expf(x));
  if (ACT == ACT_GELU)     return 0.5f * x * (1.f + erff(x * 0.70710678118654752f));
  return x;
}

// ---------------------------------------------------------------------------
// WMMA GEMM:  C[M,N] = act(A[M,K] * B[K,N] + bias[N]) + resid[M,N]
// A,B bf16 (as u16), C fp32 and/or bf16. M%64==0, K%32==0; N guarded.
// Block = 256 threads (8 waves), tile 64(M) x 64(N); wave owns one 16-wide
// N strip and two stacked 16x16 accumulators. Double-buffered LDS, one
// barrier per K-step; next tile is register-staged while WMMAs run.
// LDS row stride padded to 40 halves (80B, 16B-aligned) for bank spread.
// ---------------------------------------------------------------------------
#define LDS_STRIDE 40
#define LDS_TILE   (64 * LDS_STRIDE)

template <int ACT>
__global__ __launch_bounds__(256)
void wmma_gemm_kernel(const u16* __restrict__ A, const u16* __restrict__ B,
                      const float* __restrict__ bias, const float* __restrict__ resid,
                      float* __restrict__ Cf, u16* __restrict__ Cbf,
                      int M, int N, int K) {
  __shared__ __align__(16) u16 lA[2 * LDS_TILE];   // A tile, row-major  [m][k]
  __shared__ __align__(16) u16 lBt[2 * LDS_TILE];  // B tile, TRANSPOSED [n][k]

  const int tid   = threadIdx.x;
  const int wave  = tid >> 5;
  const int lane  = tid & 31;
  const int lhalf = lane >> 4;      // 0..1
  const int lmod  = lane & 15;      // 0..15
  const int bm = blockIdx.y * 64;
  const int bn = blockIdx.x * 64;
  const int tn  = (wave & 3) * 16;        // wave's N strip in tile
  const int tm0 = (wave >> 2) * 32;       // wave's two M rows: tm0, tm0+16
  const bool full = (bn + 64 <= N);       // uniform: no N guard needed

  v8f acc0 = {}, acc1 = {};

  // Per-lane fragment K offset inside a 32-wide K block (ISA 7.12.2):
  // lanes 0-15 hold K 0-7 & 16-23, lanes 16-31 hold K 8-15 & 24-31.
  const int kbase = lhalf ? 8 : 0;

  // A-tile staging indices (always in-bounds: M%64==0, K%32==0)
  const int arow = tid >> 2, ach = (tid & 3) * 8;
  // B-tile staging indices
  const int krow = tid >> 3, bch = (tid & 7) * 8;

  const u16* Arow = A + (size_t)(bm + arow) * K + ach;
  const u16* Brow = B + (size_t)krow * N;

  // ---- prologue: stage tile kb=0 into buffer 0 ----
  {
    *((uint4*)(lA + arow * LDS_STRIDE + ach)) = *((const uint4*)(Arow));
    if (full) {
      uint4 q = *((const uint4*)(Brow + bn + bch));
      const u16* hh = (const u16*)&q;
#pragma unroll
      for (int j = 0; j < 8; ++j) lBt[(bch + j) * LDS_STRIDE + krow] = hh[j];
    } else {
#pragma unroll
      for (int j = 0; j < 8; ++j) {
        const int gc = bn + bch + j;
        u16 hv = 0;
        if (gc < N) hv = B[(size_t)krow * N + gc];
        lBt[(bch + j) * LDS_STRIDE + krow] = hv;
      }
    }
  }
  __syncthreads();

  int cur = 0;
  for (int kb = 0; kb < K; kb += 32) {
    const bool have_next = (kb + 32 < K);
    // ---- 1. issue global loads for next tile (no wait yet) ----
    uint4 nA = {0, 0, 0, 0}, nB = {0, 0, 0, 0};
    u16 nBg[8];
    if (have_next) {
      nA = *((const uint4*)(Arow + kb + 32));
      if (full) {
        nB = *((const uint4*)(B + (size_t)(kb + 32 + krow) * N + bn + bch));
      } else {
#pragma unroll
        for (int j = 0; j < 8; ++j) {
          const int gc = bn + bch + j;
          u16 hv = 0;
          if (gc < N) hv = B[(size_t)(kb + 32 + krow) * N + gc];
          nBg[j] = hv;
        }
      }
    }

    // ---- 2. compute on current buffer: ds_load_b128 fragments + WMMA ----
    const u16* cA = lA + cur * LDS_TILE;
    const u16* cB = lBt + cur * LDS_TILE;
    union Frag { uint4 q[2]; v16u v; } fa0, fa1, fb;
    {
      const u16* ap0 = cA + (tm0 + lmod) * LDS_STRIDE + kbase;
      const u16* ap1 = cA + (tm0 + 16 + lmod) * LDS_STRIDE + kbase;
      const u16* bp  = cB + (tn + lmod) * LDS_STRIDE + kbase;
      fa0.q[0] = *((const uint4*)(ap0));
      fa0.q[1] = *((const uint4*)(ap0 + 16));
      fa1.q[0] = *((const uint4*)(ap1));
      fa1.q[1] = *((const uint4*)(ap1 + 16));
      fb.q[0]  = *((const uint4*)(bp));
      fb.q[1]  = *((const uint4*)(bp + 16));
    }
    v16bf a0 = __builtin_bit_cast(v16bf, fa0.v);
    v16bf a1 = __builtin_bit_cast(v16bf, fa1.v);
    v16bf bb = __builtin_bit_cast(v16bf, fb.v);
    acc0 = __builtin_amdgcn_wmma_f32_16x16x32_bf16(false, a0, false, bb, (short)0, acc0, false, false);
    acc1 = __builtin_amdgcn_wmma_f32_16x16x32_bf16(false, a1, false, bb, (short)0, acc1, false, false);

    // ---- 3. store staged registers into the other buffer ----
    if (have_next) {
      u16* wA = lA + (cur ^ 1) * LDS_TILE;
      u16* wB = lBt + (cur ^ 1) * LDS_TILE;
      *((uint4*)(wA + arow * LDS_STRIDE + ach)) = nA;
      if (full) {
        const u16* hh = (const u16*)&nB;
#pragma unroll
        for (int j = 0; j < 8; ++j) wB[(bch + j) * LDS_STRIDE + krow] = hh[j];
      } else {
#pragma unroll
        for (int j = 0; j < 8; ++j) wB[(bch + j) * LDS_STRIDE + krow] = nBg[j];
      }
    }
    __syncthreads();
    cur ^= 1;
  }

  // Epilogue. C/D layout: VGPR i -> M = i (+8 for lane>=16), N = lane%16.
  const int gn = bn + tn + lmod;
  if (gn < N) {
#pragma unroll
    for (int i = 0; i < 8; ++i) {
      const int m0 = bm + tm0 + i + (lhalf ? 8 : 0);
      const int m1 = m0 + 16;
      float v0 = acc0[i], v1 = acc1[i];
      if (bias) { const float bv = bias[gn]; v0 += bv; v1 += bv; }
      v0 = apply_act<ACT>(v0);
      v1 = apply_act<ACT>(v1);
      if (resid) { v0 += resid[(size_t)m0 * N + gn]; v1 += resid[(size_t)m1 * N + gn]; }
      if (Cf)  { Cf[(size_t)m0 * N + gn] = v0;        Cf[(size_t)m1 * N + gn] = v1; }
      if (Cbf) { Cbf[(size_t)m0 * N + gn] = f2bf(v0); Cbf[(size_t)m1 * N + gn] = f2bf(v1); }
    }
  }
}

// ---------------------------------------------------------------------------
// Reductions / norms
// ---------------------------------------------------------------------------
__device__ __forceinline__ float blockReduceSum(float v) {
  __shared__ float sh[8];
#pragma unroll
  for (int o = 16; o > 0; o >>= 1) v += __shfl_xor(v, o, 32);
  const int lane = threadIdx.x & 31, w = threadIdx.x >> 5;
  __syncthreads();
  if (lane == 0) sh[w] = v;
  __syncthreads();
  float tot = 0.f;
#pragma unroll
  for (int i = 0; i < 8; ++i) tot += sh[i];
  return tot;
}

__global__ __launch_bounds__(256)
void layernorm_kernel(const float* __restrict__ x, const float* __restrict__ g,
                      const float* __restrict__ bta, float* outF, u16* outBF, int D) {
  const int row = blockIdx.x;
  const float* xr = x + (size_t)row * D;
  float s = 0.f, s2 = 0.f;
  for (int i = threadIdx.x; i < D; i += 256) { const float v = xr[i]; s += v; s2 = fmaf(v, v, s2); }
  s  = blockReduceSum(s);
  s2 = blockReduceSum(s2);
  const float mean = s / D;
  const float var  = s2 / D - mean * mean;
  const float rstd = rsqrtf(var + 1e-5f);
  for (int i = threadIdx.x; i < D; i += 256) {
    const float v = (xr[i] - mean) * rstd * g[i] + bta[i];
    if (outF)  outF[(size_t)row * D + i] = v;
    if (outBF) outBF[(size_t)row * D + i] = f2bf(v);
  }
}

__global__ __launch_bounds__(256)
void rmsnorm_kernel(const float* __restrict__ x, const float* __restrict__ w,
                    u16* __restrict__ outBF, int D) {
  const int row = blockIdx.x;
  const float* xr = x + (size_t)row * D;
  float s2 = 0.f;
  for (int i = threadIdx.x; i < D; i += 256) { const float v = xr[i]; s2 = fmaf(v, v, s2); }
  s2 = blockReduceSum(s2);
  const float rms = rsqrtf(s2 / D + 1e-5f);
  for (int i = threadIdx.x; i < D; i += 256)
    outBF[(size_t)row * D + i] = f2bf(xr[i] * rms * w[i]);
}

// ---------------------------------------------------------------------------
// Elementwise / conv / scan
// ---------------------------------------------------------------------------
__global__ __launch_bounds__(256)
void f32_to_bf16_kernel(const float* __restrict__ src, u16* __restrict__ dst, size_t n) {
  const size_t i = (size_t)blockIdx.x * 256 + threadIdx.x;
  if (i < n) dst[i] = f2bf(src[i]);
}

__global__ __launch_bounds__(256)
void slice_to_bf16_kernel(const float* __restrict__ src, u16* __restrict__ dst,
                          int rows, int cols, int srcStride, int srcOff) {
  const size_t i = (size_t)blockIdx.x * 256 + threadIdx.x;
  if (i < (size_t)rows * cols) {
    const int r = (int)(i / cols), c = (int)(i % cols);
    dst[i] = f2bf(src[(size_t)r * srcStride + srcOff + c]);
  }
}

// causal depthwise conv (K=4) + bias + SiLU; xz row stride = 2*DI
__global__ __launch_bounds__(256)
void conv_silu_kernel(const float* __restrict__ xz, const float* __restrict__ w,
                      const float* __restrict__ bias, float* __restrict__ xcF,
                      u16* __restrict__ xcBF, int L, int DI, int rowStride, size_t total) {
  const size_t idx = (size_t)blockIdx.x * 256 + threadIdx.x;
  if (idx >= total) return;
  const int d = (int)(idx % DI);
  const size_t bl = idx / DI;
  const int l = (int)(bl % L);
  const int b = (int)(bl / L);
  float acc = bias[d];
#pragma unroll
  for (int k = 0; k < 4; ++k) {
    const int ll = l + k - 3;
    if (ll >= 0) acc = fmaf(w[d * 4 + k], xz[((size_t)b * L + ll) * rowStride + d], acc);
  }
  const float sv = silu_f(acc);
  xcF[idx]  = sv;
  xcBF[idx] = f2bf(sv);
}

// selective scan: 16 lanes per channel (lane = state n), 2 channels / wave
__global__ __launch_bounds__(256)
void scan_kernel(const float* __restrict__ delta, const float* __restrict__ dbc,
                 const float* __restrict__ xc, const float* __restrict__ A_log,
                 float* __restrict__ ys, int L, int DI, int G, int Boff, int Coff) {
  const int tid = threadIdx.x;
  const int lane = tid & 31;
  const int wave = tid >> 5;
  const int n = lane & 15;
  const int chan = blockIdx.x * 16 + wave * 2 + (lane >> 4);
  const int b = chan / DI;
  const int d = chan % DI;
  const float A = -__expf(A_log[(size_t)d * 16 + n]);
  float h = 0.f;
  const size_t base = (size_t)b * L;
  for (int l = 0; l < L; ++l) {
    const size_t r = base + l;
    if (l + 1 < L) __builtin_prefetch(&delta[(r + 1) * DI + d], 0, 1);
    const float dt = delta[r * DI + d];
    const float Bt = dbc[r * G + Boff + n];
    const float Ct = dbc[r * G + Coff + n];
    const float u  = xc[r * DI + d];
    h = __expf(dt * A) * h + dt * Bt * u;
    float y = h * Ct;
    y += __shfl_xor(y, 1, 16);
    y += __shfl_xor(y, 2, 16);
    y += __shfl_xor(y, 4, 16);
    y += __shfl_xor(y, 8, 16);
    if (n == 0) ys[r * DI + d] = y;
  }
}

// yg = bf16( (ys + xc*D) * silu(z) ),  z = xz[:, DI:2DI]
__global__ __launch_bounds__(256)
void gate_kernel(const float* __restrict__ ys, const float* __restrict__ xc,
                 const float* __restrict__ Dp, const float* __restrict__ xz,
                 u16* __restrict__ yg, int DI, int rowStride, size_t total) {
  const size_t idx = (size_t)blockIdx.x * 256 + threadIdx.x;
  if (idx >= total) return;
  const int d = (int)(idx % DI);
  const size_t row = idx / DI;
  const float z = xz[row * rowStride + DI + d];
  const float y = fmaf(xc[idx], Dp[d], ys[idx]) * silu_f(z);
  yg[idx] = f2bf(y);
}

// h2 = u*v + x
__global__ __launch_bounds__(256)
void h2_kernel(const float* __restrict__ u, const float* __restrict__ v,
               const float* __restrict__ x, float* __restrict__ h2, size_t total) {
  const size_t i = (size_t)blockIdx.x * 256 + threadIdx.x;
  if (i < total) h2[i] = fmaf(u[i], v[i], x[i]);
}

// ---------------------------------------------------------------------------
// Host
// ---------------------------------------------------------------------------
static inline unsigned nblk(size_t n) { return (unsigned)((n + 255) / 256); }

static void launch_gemm(int act, const u16* A, const u16* B, const float* bias,
                        const float* resid, float* Cf, u16* Cbf,
                        int M, int N, int K, hipStream_t s) {
  dim3 g((N + 63) / 64, M / 64), b(256);
  switch (act) {
    case ACT_SOFTPLUS:
      wmma_gemm_kernel<ACT_SOFTPLUS><<<g, b, 0, s>>>(A, B, bias, resid, Cf, Cbf, M, N, K); break;
    case ACT_GELU:
      wmma_gemm_kernel<ACT_GELU><<<g, b, 0, s>>>(A, B, bias, resid, Cf, Cbf, M, N, K); break;
    default:
      wmma_gemm_kernel<ACT_NONE><<<g, b, 0, s>>>(A, B, bias, resid, Cf, Cbf, M, N, K); break;
  }
}

extern "C" void kernel_launch(void* const* d_in, const int* in_sizes, int n_in,
                              void* d_out, int out_size, void* d_ws, size_t ws_size,
                              hipStream_t stream) {
  (void)in_sizes; (void)n_in; (void)out_size; (void)ws_size;
  const int Bt = 2, L = 2048, DM = 1024, DI = 2048, N2 = 4096, G = 96, DT = 64, FF = 4096;
  const int M = Bt * L;  // 4096

  const float* x     = (const float*)d_in[0];
  const float* ln_g  = (const float*)d_in[1];
  const float* ln_b  = (const float*)d_in[2];
  const float* fw1   = (const float*)d_in[27];
  const float* fb1   = (const float*)d_in[28];
  const float* fw2   = (const float*)d_in[29];
  const float* fb2   = (const float*)d_in[30];

  // workspace arena (256B aligned slots)
  char* ws = (char*)d_ws;
  size_t off = 0;
  auto alloc = [&](size_t bytes) -> void* {
    void* p = ws + off;
    off += (bytes + 255) & ~(size_t)255;
    return p;
  };
  float* h_f     = (float*)alloc((size_t)M * DM * 4);
  float* u_f     = (float*)alloc((size_t)M * DM * 4);
  float* v_f     = (float*)alloc((size_t)M * DM * 4);
  float* h2_f    = (float*)alloc((size_t)M * DM * 4);
  u16*   hrms_bf = (u16*)  alloc((size_t)M * DM * 2);
  u16*   h2ln_bf = (u16*)  alloc((size_t)M * DM * 2);
  u16*   ff1_bf  = (u16*)  alloc((size_t)M * FF * 2);
  float* xz_f    = (float*)alloc((size_t)M * N2 * 4);
  float* xc_f    = (float*)alloc((size_t)M * DI * 4);
  u16*   xc_bf   = (u16*)  alloc((size_t)M * DI * 2);
  float* dbc_f   = (float*)alloc((size_t)M * G * 4);
  u16*   dlt_bf  = (u16*)  alloc((size_t)M * DT * 2);
  float* delta_f = (float*)alloc((size_t)M * DI * 4);
  float* ys_f    = (float*)alloc((size_t)M * DI * 4);
  u16*   yg_bf   = (u16*)  alloc((size_t)M * DI * 2);
  u16*   w_in    = (u16*)  alloc((size_t)DM * N2 * 2);
  u16*   w_xp    = (u16*)  alloc((size_t)DI * G * 2);
  u16*   w_dt    = (u16*)  alloc((size_t)DT * DI * 2);
  u16*   w_out   = (u16*)  alloc((size_t)DI * DM * 2);
  u16*   w_f1    = (u16*)  alloc((size_t)DM * FF * 2);
  u16*   w_f2    = (u16*)  alloc((size_t)FF * DM * 2);

  auto cvt = [&](const float* s, u16* d, size_t n) {
    f32_to_bf16_kernel<<<nblk(n), 256, 0, stream>>>(s, d, n);
  };

  // h = LN(x)
  layernorm_kernel<<<M, 256, 0, stream>>>(x, ln_g, ln_b, h_f, nullptr, DM);

  for (int br = 0; br < 2; ++br) {
    const int p = 3 + br * 12;
    const float* rms_w   = (const float*)d_in[p + 0];
    const float* in_w    = (const float*)d_in[p + 1];
    const float* in_b    = (const float*)d_in[p + 2];
    const float* conv_w  = (const float*)d_in[p + 3];
    const float* conv_b  = (const float*)d_in[p + 4];
    const float* xproj_w = (const float*)d_in[p + 5];
    const float* dt_w    = (const float*)d_in[p + 6];
    const float* dt_b    = (const float*)d_in[p + 7];
    const float* A_log   = (const float*)d_in[p + 8];
    const float* Dp      = (const float*)d_in[p + 9];
    const float* out_w   = (const float*)d_in[p + 10];
    const float* out_b   = (const float*)d_in[p + 11];

    rmsnorm_kernel<<<M, 256, 0, stream>>>(h_f, rms_w, hrms_bf, DM);
    cvt(in_w,    w_in,  (size_t)DM * N2);
    cvt(xproj_w, w_xp,  (size_t)DI * G);
    cvt(dt_w,    w_dt,  (size_t)DT * DI);
    cvt(out_w,   w_out, (size_t)DI * DM);

    // xz = hrms @ in_w + in_b
    launch_gemm(ACT_NONE, hrms_bf, w_in, in_b, nullptr, xz_f, nullptr, M, N2, DM, stream);
    // causal depthwise conv + SiLU
    conv_silu_kernel<<<nblk((size_t)M * DI), 256, 0, stream>>>(
        xz_f, conv_w, conv_b, xc_f, xc_bf, L, DI, N2, (size_t)M * DI);
    // dbc = xc @ xproj_w (no bias, N=96)
    launch_gemm(ACT_NONE, xc_bf, w_xp, nullptr, nullptr, dbc_f, nullptr, M, G, DI, stream);
    // dlt = dbc[:, :64] -> bf16
    slice_to_bf16_kernel<<<nblk((size_t)M * DT), 256, 0, stream>>>(dbc_f, dlt_bf, M, DT, G, 0);
    // delta = softplus(dlt @ dt_w + dt_b)
    launch_gemm(ACT_SOFTPLUS, dlt_bf, w_dt, dt_b, nullptr, delta_f, nullptr, M, DI, DT, stream);
    // selective scan
    scan_kernel<<<Bt * DI / 16, 256, 0, stream>>>(
        delta_f, dbc_f, xc_f, A_log, ys_f, L, DI, G, DT, DT + 16);
    // yg = (ys + xc*D) * silu(z)
    gate_kernel<<<nblk((size_t)M * DI), 256, 0, stream>>>(
        ys_f, xc_f, Dp, xz_f, yg_bf, DI, N2, (size_t)M * DI);
    // branch out = yg @ out_w + out_b + h   (mamba residual)
    launch_gemm(ACT_NONE, yg_bf, w_out, out_b, h_f, (br ? v_f : u_f), nullptr, M, DM, DI, stream);
  }

  // h2 = u*v + x ; second LN (shared weights) -> bf16
  h2_kernel<<<nblk((size_t)M * DM), 256, 0, stream>>>(u_f, v_f, x, h2_f, (size_t)M * DM);
  layernorm_kernel<<<M, 256, 0, stream>>>(h2_f, ln_g, ln_b, nullptr, h2ln_bf, DM);

  cvt(fw1, w_f1, (size_t)DM * FF);
  cvt(fw2, w_f2, (size_t)FF * DM);
  // ff1 = gelu(h2ln @ w1 + b1) -> bf16 only
  launch_gemm(ACT_GELU, h2ln_bf, w_f1, fb1, nullptr, nullptr, ff1_bf, M, FF, DM, stream);
  // out = ff1 @ w2 + b2 + h2
  launch_gemm(ACT_NONE, ff1_bf, w_f2, fb2, h2_f, (float*)d_out, nullptr, M, DM, FF, stream);
}
